// MultiHeadAttention_62234076119080
// MI455X (gfx1250) — matrix operand order repaired and checked by reading, not compiled, not run on hardware
//
#include <hip/hip_runtime.h>

// ---------------------------------------------------------------------------
// Multi-head attention for gfx1250 (MI455X).
// f16 WMMA + flash attention + async global->LDS staging (double buffered).
// Fixed problem: B=4, S=2048, D=1024, H=16, Hd=64.
// ---------------------------------------------------------------------------

typedef _Float16 v16h __attribute__((ext_vector_type(16)));
typedef _Float16 v8h  __attribute__((ext_vector_type(8)));
typedef _Float16 v4h  __attribute__((ext_vector_type(4)));
typedef float    v8f  __attribute__((ext_vector_type(8)));
typedef float    v4f  __attribute__((ext_vector_type(4)));

constexpr int D_MODEL = 1024;
constexpr int NHEAD   = 16;
constexpr int HDIM    = 64;
constexpr int BATCH   = 4;
constexpr int SEQ     = 2048;
constexpr int MTOT    = BATCH * SEQ;   // 8192

#define WMMA_F16(a, b, c) \
  __builtin_amdgcn_wmma_f32_16x16x32_f16(false, (a), false, (b), (short)0, (c), false, false)

__device__ inline v8f zero8() {
  v8f z;
#pragma unroll
  for (int i = 0; i < 8; ++i) z[i] = 0.0f;
  return z;
}

// Async DMA: 16 bytes global -> LDS, tracked by ASYNCcnt (per-lane addresses).
__device__ inline void async_b128(void* lds_dst, const void* gsrc) {
  unsigned l = (unsigned)(size_t)lds_dst;   // low 32 bits of generic addr = LDS offset
  asm volatile("global_load_async_to_lds_b128 %0, %1, off"
               :: "v"(l), "v"(gsrc) : "memory");
}
#define ASYNC_WAIT(n) asm volatile("s_wait_asynccnt " #n ::: "memory")

// A-matrix 16x32 f16 fragment from row-major storage, row stride `ld` halves.
// ISA layout: lane L -> row m=L&15; g=L>>4; h[0..7]=K(g*8..g*8+7), h[8..15]=K(16+g*8..).
__device__ inline v16h load_frag_a(const _Float16* __restrict__ p, int ld, int lane) {
  const int m = lane & 15, g = lane >> 4;
  union { v16h v; v8h h[2]; } u;
  const _Float16* r = p + (size_t)m * ld;
  u.h[0] = *(const v8h*)(r + g * 8);
  u.h[1] = *(const v8h*)(r + 16 + g * 8);
  return u.v;
}

// B-matrix 32x16 f16 fragment. Source is "N-major": row n holds 32 contiguous K values.
// ISA layout: lane L -> col n=L&15; g=L>>4; h[0..7]=K(g*8..g*8+7), h[8..15]=K(16+g*8..).
__device__ inline v16h load_frag_b(const _Float16* __restrict__ p, int ld, int lane) {
  const int n = lane & 15, g = lane >> 4;
  union { v16h v; v8h h[2]; } u;
  const _Float16* r = p + (size_t)n * ld + g * 8;
  u.h[0] = *(const v8h*)(r);
  u.h[1] = *(const v8h*)(r + 16);
  return u.v;
}

// ---------------------------------------------------------------------------
// Kernel 0: one-time f32 -> f16 conversion (x and the four weight matrices).
// ---------------------------------------------------------------------------
__global__ __launch_bounds__(256) void cvt_f32_f16_kernel(
    const float* __restrict__ src, _Float16* __restrict__ dst, int n4)
{
  const int i = blockIdx.x * 256 + threadIdx.x;
  if (i < n4) {
    v4f a = *(const v4f*)(src + (size_t)i * 4);
    v4h h;
#pragma unroll
    for (int e = 0; e < 4; ++e) h[e] = (_Float16)a[e];
    *(v4h*)(dst + (size_t)i * 4) = h;
  }
}

// ---------------------------------------------------------------------------
// Kernel 1: fused QKV projection.  C = x @ W^T + b for Wq, Wk, Wv (all f16 in).
// Block tile 128(M) x 64(N), 8 waves (4x2), each wave 32x32 = 2x2 WMMA tiles.
// Async double-buffered staging: 5 b128 copies / thread / stage.
// Q, K stored f16 row-major [MTOT][D].  V stored transposed: Vt[b*D + n][s].
// ---------------------------------------------------------------------------
__global__ __launch_bounds__(256) void qkv_proj_kernel(
    const _Float16* __restrict__ xh,
    const _Float16* __restrict__ wqh, const float* __restrict__ bq,
    const _Float16* __restrict__ wkh, const float* __restrict__ bk,
    const _Float16* __restrict__ wvh, const float* __restrict__ bv,
    _Float16* __restrict__ Qo, _Float16* __restrict__ Ko,
    _Float16* __restrict__ Vt)
{
  __shared__ __align__(16) _Float16 sA[2][128][32];     // 16 KB
  __shared__ __align__(16) _Float16 sW[2][3][64][32];   // 24 KB

  const int tid  = threadIdx.x;
  const int lane = tid & 31;
  const int wave = tid >> 5;
  const int wm   = wave & 3;
  const int wn   = wave >> 2;
  const int tileM = blockIdx.x * 128;
  const int tileN = blockIdx.y * 64;
  const _Float16* wmat[3] = { wqh, wkh, wvh };

  // 16B chunk coordinates (8 halves): A tile has 4 chunks/row, W tiles 4 chunks/row.
  const int ar = tid >> 2;          // 0..63
  const int ac = (tid & 3) * 8;

  auto issue = [&](int stage) {
    const int buf = stage & 1;
    const int k0  = stage * 32;
    async_b128(&sA[buf][ar][ac],
               xh + (size_t)(tileM + ar) * D_MODEL + k0 + ac);
    async_b128(&sA[buf][ar + 64][ac],
               xh + (size_t)(tileM + ar + 64) * D_MODEL + k0 + ac);
#pragma unroll
    for (int p = 0; p < 3; ++p)
      async_b128(&sW[buf][p][ar][ac],
                 wmat[p] + (size_t)(tileN + ar) * D_MODEL + k0 + ac);
  };

  v8f acc[3][2][2];
#pragma unroll
  for (int p = 0; p < 3; ++p)
#pragma unroll
    for (int i = 0; i < 2; ++i)
#pragma unroll
      for (int j = 0; j < 2; ++j) acc[p][i][j] = zero8();

  constexpr int NSTAGE = D_MODEL / 32;   // 32
  issue(0);
  for (int s = 0; s < NSTAGE; ++s) {
    if (s + 1 < NSTAGE) {
      issue(s + 1);
      ASYNC_WAIT(5);     // drain stage s, keep stage s+1 in flight
    } else {
      ASYNC_WAIT(0);
    }
    __syncthreads();

    const int buf = s & 1;
    v16h af[2];
    af[0] = load_frag_a(&sA[buf][wm * 32][0], 32, lane);
    af[1] = load_frag_a(&sA[buf][wm * 32 + 16][0], 32, lane);
#pragma unroll
    for (int p = 0; p < 3; ++p) {
#pragma unroll
      for (int sn = 0; sn < 2; ++sn) {
        v16h bf = load_frag_b(&sW[buf][p][wn * 32 + sn * 16][0], 32, lane);
        acc[p][0][sn] = WMMA_F16(af[0], bf, acc[p][0][sn]);
        acc[p][1][sn] = WMMA_F16(af[1], bf, acc[p][1][sn]);
      }
    }
    __syncthreads();   // reads done before stage s+2 overwrites buf
  }

  // epilogue: bias + store.  C layout: lane L, vgpr r -> row r+8*(L>>4), col L&15.
  const int m = lane & 15, g = lane >> 4;
#pragma unroll
  for (int sm = 0; sm < 2; ++sm) {
#pragma unroll
    for (int sn = 0; sn < 2; ++sn) {
      const int col  = tileN + wn * 32 + sn * 16 + m;
      const int rowb = tileM + wm * 32 + sm * 16 + 8 * g;
      const float biq = bq[col], bik = bk[col], biv = bv[col];
#pragma unroll
      for (int r = 0; r < 8; ++r) {
        const int row = rowb + r;
        Qo[(size_t)row * D_MODEL + col] = (_Float16)(acc[0][sm][sn][r] + biq);
        Ko[(size_t)row * D_MODEL + col] = (_Float16)(acc[1][sm][sn][r] + bik);
        const int bb = row >> 11;        // row / SEQ
        const int ss = row & (SEQ - 1);  // row % SEQ
        Vt[((size_t)(bb * D_MODEL + col)) * SEQ + ss] =
            (_Float16)(acc[2][sm][sn][r] + biv);
      }
    }
  }
}

// ---------------------------------------------------------------------------
// Kernel 2: flash attention.  4 waves/block, each wave owns one 16-row q-tile
// of one (b,h).  K/V tiles (64 keys) staged ONCE per block via async DMA into
// LDS (double buffered) and shared by all 4 waves; online softmax; P goes
// through a wave-private LDS slice to convert C-layout -> A-layout.
// ---------------------------------------------------------------------------
__global__ __launch_bounds__(128) void attn_kernel(
    const _Float16* __restrict__ Q, const _Float16* __restrict__ Kf,
    const _Float16* __restrict__ Vt, _Float16* __restrict__ O)
{
  __shared__ __align__(16) _Float16 sK[2][64][64];   // 16 KB
  __shared__ __align__(16) _Float16 sV[2][64][64];   // 16 KB
  __shared__ __align__(16) _Float16 sP[4][16][64];   //  8 KB (wave-private)

  const int tid  = threadIdx.x;
  const int lane = tid & 31;
  const int wave = tid >> 5;
  const int m = lane & 15, g = lane >> 4;
  const int bh = blockIdx.x;            // 0..63
  const int b  = bh >> 4, h = bh & 15;
  const int q0 = blockIdx.y * 64 + wave * 16;

  const _Float16* qbase = Q  + ((size_t)(b * SEQ + q0)) * D_MODEL + h * HDIM;
  const _Float16* kbase = Kf + ((size_t)(b * SEQ)) * D_MODEL + h * HDIM;
  const _Float16* vbase = Vt + ((size_t)(b * D_MODEL + h * HDIM)) * SEQ;
  _Float16*       obase = O  + ((size_t)(b * SEQ + q0)) * D_MODEL + h * HDIM;

  // K/V tile staging: 8 KB each = 512 16B-chunks; 128 threads x 4 chunks.
  auto issue = [&](int stage) {
    const int buf = stage & 1;
    const int kt  = stage * 64;
#pragma unroll
    for (int i = 0; i < 4; ++i) {
      const int ch  = tid + 128 * i;     // 0..511
      const int row = ch >> 3;           // 0..63
      const int col = (ch & 7) * 8;      // 0..56
      async_b128(&sK[buf][row][col],
                 kbase + (size_t)(kt + row) * D_MODEL + col);
      async_b128(&sV[buf][row][col],
                 vbase + (size_t)row * SEQ + kt + col);
    }
  };

  v16h aq[2];
  aq[0] = load_frag_a(qbase, D_MODEL, lane);
  aq[1] = load_frag_a(qbase + 32, D_MODEL, lane);

  v8f oacc[4];
#pragma unroll
  for (int dt = 0; dt < 4; ++dt) oacc[dt] = zero8();
  float mrow[8], lrow[8];
#pragma unroll
  for (int r = 0; r < 8; ++r) { mrow[r] = -1e30f; lrow[r] = 0.0f; }

  const float scale = 0.125f;            // 1/sqrt(64)
  constexpr int NSTAGE = SEQ / 64;       // 32

  issue(0);
  for (int it = 0; it < NSTAGE; ++it) {
    if (it + 1 < NSTAGE) {
      issue(it + 1);
      ASYNC_WAIT(8);
    } else {
      ASYNC_WAIT(0);
    }
    __syncthreads();
    const int buf = it & 1;

    // ---- scores S = scale * Q Kt : 16x64 in 4 accumulators ----
    v8f sacc[4];
#pragma unroll
    for (int nt = 0; nt < 4; ++nt) {
      v8f a = zero8();
      a = WMMA_F16(aq[0], load_frag_b(&sK[buf][nt * 16][0],  64, lane), a);
      a = WMMA_F16(aq[1], load_frag_b(&sK[buf][nt * 16][32], 64, lane), a);
#pragma unroll
      for (int r = 0; r < 8; ++r) a[r] *= scale;
      sacc[nt] = a;
    }
    // ---- online softmax: row max (butterfly over 16-lane group) ----
    float arow[8];
#pragma unroll
    for (int r = 0; r < 8; ++r) {
      float t = fmaxf(fmaxf(sacc[0][r], sacc[1][r]), fmaxf(sacc[2][r], sacc[3][r]));
      t = fmaxf(t, __shfl_xor(t, 1, 32));
      t = fmaxf(t, __shfl_xor(t, 2, 32));
      t = fmaxf(t, __shfl_xor(t, 4, 32));
      t = fmaxf(t, __shfl_xor(t, 8, 32));
      const float mn = fmaxf(mrow[r], t);
      arow[r] = __expf(mrow[r] - mn);
      mrow[r] = mn;
      lrow[r] *= arow[r];
    }
    // ---- p = exp(s - m), row sums ----
    float rs[8];
#pragma unroll
    for (int r = 0; r < 8; ++r) rs[r] = 0.0f;
#pragma unroll
    for (int nt = 0; nt < 4; ++nt) {
#pragma unroll
      for (int r = 0; r < 8; ++r) {
        const float p = __expf(sacc[nt][r] - mrow[r]);
        sacc[nt][r] = p;
        rs[r] += p;
      }
    }
#pragma unroll
    for (int r = 0; r < 8; ++r) {
      float t = rs[r];
      t += __shfl_xor(t, 1, 32);
      t += __shfl_xor(t, 2, 32);
      t += __shfl_xor(t, 4, 32);
      t += __shfl_xor(t, 8, 32);
      lrow[r] += t;
    }
    // ---- rescale running output ----
#pragma unroll
    for (int dt = 0; dt < 4; ++dt)
#pragma unroll
      for (int r = 0; r < 8; ++r) oacc[dt][r] *= arow[r];

    // ---- P: C-layout -> row-major LDS (wave-private; DS ops are in-order) ----
#pragma unroll
    for (int nt = 0; nt < 4; ++nt)
#pragma unroll
      for (int r = 0; r < 8; ++r)
        sP[wave][r + 8 * g][nt * 16 + m] = (_Float16)sacc[nt][r];

    v16h ap0 = load_frag_a(&sP[wave][0][0],  64, lane);
    v16h ap1 = load_frag_a(&sP[wave][0][32], 64, lane);

    // ---- O += P V  (B-fragments contiguous thanks to transposed Vt) ----
#pragma unroll
    for (int dt = 0; dt < 4; ++dt) {
      oacc[dt] = WMMA_F16(ap0, load_frag_b(&sV[buf][dt * 16][0],  64, lane), oacc[dt]);
      oacc[dt] = WMMA_F16(ap1, load_frag_b(&sV[buf][dt * 16][32], 64, lane), oacc[dt]);
    }
    __syncthreads();   // all waves done with buf before stage it+2 overwrites it
  }

  // ---- normalize and store [B,S,D] f16 ----
#pragma unroll
  for (int dt = 0; dt < 4; ++dt) {
#pragma unroll
    for (int r = 0; r < 8; ++r) {
      const float v = oacc[dt][r] / lrow[r];
      obase[(size_t)(r + 8 * g) * D_MODEL + dt * 16 + m] = (_Float16)v;
    }
  }
}

// ---------------------------------------------------------------------------
// Kernel 3: output projection.  out = attn(f16) @ Wo^T(f16) + bo, fp32 result.
// Same tiling as kernel 1; async double-buffered staging (3 copies/thread).
// ---------------------------------------------------------------------------
__global__ __launch_bounds__(256) void out_proj_kernel(
    const _Float16* __restrict__ Ain,
    const _Float16* __restrict__ woh, const float* __restrict__ bo,
    float* __restrict__ out)
{
  __shared__ __align__(16) _Float16 sA[2][128][32];   // 16 KB
  __shared__ __align__(16) _Float16 sW[2][64][32];    //  8 KB

  const int tid  = threadIdx.x;
  const int lane = tid & 31;
  const int wave = tid >> 5;
  const int wm   = wave & 3;
  const int wn   = wave >> 2;
  const int tileM = blockIdx.x * 128;
  const int tileN = blockIdx.y * 64;

  const int ar = tid >> 2;
  const int ac = (tid & 3) * 8;

  auto issue = [&](int stage) {
    const int buf = stage & 1;
    const int k0  = stage * 32;
    async_b128(&sA[buf][ar][ac],
               Ain + (size_t)(tileM + ar) * D_MODEL + k0 + ac);
    async_b128(&sA[buf][ar + 64][ac],
               Ain + (size_t)(tileM + ar + 64) * D_MODEL + k0 + ac);
    async_b128(&sW[buf][ar][ac],
               woh + (size_t)(tileN + ar) * D_MODEL + k0 + ac);
  };

  v8f acc[2][2];
#pragma unroll
  for (int i = 0; i < 2; ++i)
#pragma unroll
    for (int j = 0; j < 2; ++j) acc[i][j] = zero8();

  constexpr int NSTAGE = D_MODEL / 32;
  issue(0);
  for (int s = 0; s < NSTAGE; ++s) {
    if (s + 1 < NSTAGE) {
      issue(s + 1);
      ASYNC_WAIT(3);
    } else {
      ASYNC_WAIT(0);
    }
    __syncthreads();

    const int buf = s & 1;
    v16h af[2];
    af[0] = load_frag_a(&sA[buf][wm * 32][0], 32, lane);
    af[1] = load_frag_a(&sA[buf][wm * 32 + 16][0], 32, lane);
#pragma unroll
    for (int sn = 0; sn < 2; ++sn) {
      v16h bf = load_frag_b(&sW[buf][wn * 32 + sn * 16][0], 32, lane);
      acc[0][sn] = WMMA_F16(af[0], bf, acc[0][sn]);
      acc[1][sn] = WMMA_F16(af[1], bf, acc[1][sn]);
    }
    __syncthreads();
  }

  const int m = lane & 15, g = lane >> 4;
#pragma unroll
  for (int sm = 0; sm < 2; ++sm) {
#pragma unroll
    for (int sn = 0; sn < 2; ++sn) {
      const int col  = tileN + wn * 32 + sn * 16 + m;
      const int rowb = tileM + wm * 32 + sm * 16 + 8 * g;
      const float bi = bo[col];
#pragma unroll
      for (int r = 0; r < 8; ++r)
        out[(size_t)(rowb + r) * D_MODEL + col] = acc[sm][sn][r] + bi;
    }
  }
}

// ---------------------------------------------------------------------------
extern "C" void kernel_launch(void* const* d_in, const int* in_sizes, int n_in,
                              void* d_out, int out_size, void* d_ws, size_t ws_size,
                              hipStream_t stream) {
  (void)in_sizes; (void)n_in; (void)out_size; (void)ws_size;
  const float* x  = (const float*)d_in[0];
  const float* wq = (const float*)d_in[1];
  const float* bq = (const float*)d_in[2];
  const float* wk = (const float*)d_in[3];
  const float* bk = (const float*)d_in[4];
  const float* wv = (const float*)d_in[5];
  const float* bv = (const float*)d_in[6];
  const float* wo = (const float*)d_in[7];
  const float* bo = (const float*)d_in[8];

  char* ws = (char*)d_ws;
  const size_t SZA = (size_t)MTOT * D_MODEL * sizeof(_Float16);      // 16 MB
  const size_t SZW = (size_t)D_MODEL * D_MODEL * sizeof(_Float16);   //  2 MB
  _Float16* Q   = (_Float16*)(ws);
  _Float16* K   = (_Float16*)(ws + SZA);
  _Float16* Vt  = (_Float16*)(ws + 2 * SZA);
  _Float16* AO  = (_Float16*)(ws + 3 * SZA);
  _Float16* xh  = (_Float16*)(ws + 4 * SZA);
  _Float16* wqh = (_Float16*)(ws + 5 * SZA);
  _Float16* wkh = (_Float16*)(ws + 5 * SZA + SZW);
  _Float16* wvh = (_Float16*)(ws + 5 * SZA + 2 * SZW);
  _Float16* woh = (_Float16*)(ws + 5 * SZA + 3 * SZW);

  // one-time f32 -> f16 packing
  const int nx4 = MTOT * D_MODEL / 4;        // 2M
  const int nw4 = D_MODEL * D_MODEL / 4;     // 256K
  cvt_f32_f16_kernel<<<(nx4 + 255) / 256, 256, 0, stream>>>(x,  xh,  nx4);
  cvt_f32_f16_kernel<<<(nw4 + 255) / 256, 256, 0, stream>>>(wq, wqh, nw4);
  cvt_f32_f16_kernel<<<(nw4 + 255) / 256, 256, 0, stream>>>(wk, wkh, nw4);
  cvt_f32_f16_kernel<<<(nw4 + 255) / 256, 256, 0, stream>>>(wv, wvh, nw4);
  cvt_f32_f16_kernel<<<(nw4 + 255) / 256, 256, 0, stream>>>(wo, woh, nw4);

  dim3 gProj(MTOT / 128, D_MODEL / 64);   // 64 x 16
  qkv_proj_kernel<<<gProj, 256, 0, stream>>>(xh, wqh, bq, wkh, bk, wvh, bv, Q, K, Vt);

  dim3 gAttn(BATCH * NHEAD, SEQ / 64);    // 64 x 32
  attn_kernel<<<gAttn, 128, 0, stream>>>(Q, K, Vt, AO);

  out_proj_kernel<<<gProj, 256, 0, stream>>>(AO, woh, bo, (float*)d_out);
}
